// MultiHeadAttention_27693949125393
// MI455X (gfx1250) — matrix operand order repaired and checked by reading, not compiled, not run on hardware
//
#include <hip/hip_runtime.h>
#include <hip/hip_bf16.h>

// ---------------------------------------------------------------------------
// MHA forward for MI455X (gfx1250): bf16 WMMA (16x16x32) everywhere, f32 acc.
// GEMMs: 128x128 tiles, double-buffered async global->LDS staging (ASYNCcnt).
// ---------------------------------------------------------------------------

typedef __attribute__((ext_vector_type(16))) __bf16        v16bf;
typedef __attribute__((ext_vector_type(8)))  float         v8f;
typedef __attribute__((ext_vector_type(4)))  float         v4f;
typedef __attribute__((ext_vector_type(4)))  unsigned int  v4u;
typedef __attribute__((ext_vector_type(2)))  unsigned int  v2u;

union FragAB {               // 32 bytes = one WMMA 16-bit A/B operand per lane
    v4u            q[2];
    v16bf          v;
    unsigned short s[16];
};

static constexpr int kB  = 2;
static constexpr int kS  = 2048;
static constexpr int kD  = 1024;
static constexpr int kH  = 16;
static constexpr int kDK = 64;          // kD / kH

__device__ __forceinline__ unsigned short f2bf(float f) {
    unsigned int u = __float_as_uint(f);
    unsigned int r = u + 0x7FFFu + ((u >> 16) & 1u);   // round-to-nearest-even
    return (unsigned short)(r >> 16);
}

// Output modes for the GEMM epilogue
#define OUT_BF16   0   // bf16, row-major (M x N)
#define OUT_F32NT  1   // f32, row-major, non-temporal store
#define OUT_VT     2   // bf16, transposed per head: Vt[(b*H+h)*DK+dl][s]

// ---------------------------------------------------------------------------
// Prep pass 1: f32 -> bf16 conversion of the three activation tensors.
// ---------------------------------------------------------------------------
__global__ __launch_bounds__(256)
void cvtBf16(const float* __restrict__ A0, const float* __restrict__ A1,
             const float* __restrict__ A2,
             unsigned short* __restrict__ O0, unsigned short* __restrict__ O1,
             unsigned short* __restrict__ O2)
{
    const float*    src = (blockIdx.z == 0) ? A0 : (blockIdx.z == 1) ? A1 : A2;
    unsigned short* dst = (blockIdx.z == 0) ? O0 : (blockIdx.z == 1) ? O1 : O2;
    size_t i = ((size_t)blockIdx.x * 256 + threadIdx.x) * 4;
    v4f x = *(const v4f*)(src + i);
    union { unsigned short s[4]; v2u u; } p;
    p.s[0] = f2bf(x[0]); p.s[1] = f2bf(x[1]);
    p.s[2] = f2bf(x[2]); p.s[3] = f2bf(x[3]);
    *(v2u*)(dst + i) = p.u;
}

// ---------------------------------------------------------------------------
// Prep pass 2: W (K x N, f32) -> Wt (N x K, bf16) for all four weights.
// 64x64 tiles through LDS; both global sides coalesced.
// ---------------------------------------------------------------------------
__global__ __launch_bounds__(256)
void transposeW(const float* __restrict__ W0, const float* __restrict__ W1,
                const float* __restrict__ W2, const float* __restrict__ W3,
                unsigned short* __restrict__ T0, unsigned short* __restrict__ T1,
                unsigned short* __restrict__ T2, unsigned short* __restrict__ T3)
{
    __shared__ unsigned short tile[64 * 68];
    const float*    W = (blockIdx.z == 0) ? W0 : (blockIdx.z == 1) ? W1
                      : (blockIdx.z == 2) ? W2 : W3;
    unsigned short* T = (blockIdx.z == 0) ? T0 : (blockIdx.z == 1) ? T1
                      : (blockIdx.z == 2) ? T2 : T3;
    const int t  = threadIdx.x;
    const int k0 = blockIdx.x * 64, n0 = blockIdx.y * 64;
    for (int idx = t; idx < 64 * 64; idx += 256) {
        int r = idx >> 6, c = idx & 63;                 // r = k, c = n
        tile[c * 68 + r] = f2bf(W[(size_t)(k0 + r) * kD + n0 + c]);
    }
    __syncthreads();
    for (int idx = t; idx < 64 * 64; idx += 256) {
        int r = idx >> 6, c = idx & 63;                 // r = n, c = k
        T[(size_t)(n0 + r) * kD + k0 + c] = tile[r * 68 + c];
    }
}

// ---------------------------------------------------------------------------
// 128x128-tile GEMM:  O[M,N] = X[M,K] @ Wt[N,K]^T   (both operands bf16)
// 256 threads = 8 waves in a 4(M) x 2(N) grid; each wave: 32 rows x 64 cols
// = 2x4 accumulators, 8 WMMAs per 32-wide k-step.
// Double-buffered: tile i+1 is staged with global_load_async_to_lds_b128
// while tile i is consumed by WMMAs; one barrier per k-step.
// ---------------------------------------------------------------------------
template<int OUT_MODE>
__global__ __launch_bounds__(256)
void gemm128(const unsigned short* __restrict__ X,
             const unsigned short* __restrict__ Wt,
             void* __restrict__ Ov, int M, int N, int K)
{
    constexpr int XS = 40;   // padded row stride (ushorts): 80B, 16B-aligned
    __shared__ unsigned short ldsX[2][128 * XS];   // X tiles  [row][k]
    __shared__ unsigned short ldsW[2][128 * XS];   // Wt tiles [n][k]

    const int t    = threadIdx.x;
    const int lane = t & 31;
    const int w    = t >> 5;
    const int wr   = w & 3;              // 4 row-groups of 32
    const int wc   = w >> 2;             // 2 col-groups of 64
    const int hi   = (lane >= 16) ? 1 : 0;
    const int tM   = blockIdx.y * 128;
    const int tN   = blockIdx.x * 128;

    // async stage of one 128x32 tile pair: 1024 b128 lane-transfers total
    auto stage = [&](int buf, int k0) {
        #pragma unroll
        for (int rep = 0; rep < 2; ++rep) {
            int idx = t + rep * 256;
            int r = idx >> 2, seg = idx & 3;
            unsigned lx = (unsigned)(uintptr_t)&ldsX[buf][r * XS + seg * 8];
            unsigned lw = (unsigned)(uintptr_t)&ldsW[buf][r * XS + seg * 8];
            const unsigned short* gx = X  + (size_t)(tM + r) * K + k0 + seg * 8;
            const unsigned short* gw = Wt + (size_t)(tN + r) * K + k0 + seg * 8;
            asm volatile("global_load_async_to_lds_b128 %0, %1, off"
                         :: "v"(lx), "v"(gx) : "memory");
            asm volatile("global_load_async_to_lds_b128 %0, %1, off"
                         :: "v"(lw), "v"(gw) : "memory");
        }
    };

    v8f acc[2][4] = {};
    const int nSteps = K / 32;
    stage(0, 0);

    for (int it = 0; it < nSteps; ++it) {
        const int cur = it & 1;
        // complete the stage issued one iteration ago (overlapped with WMMAs)
        asm volatile("s_wait_asynccnt 0x0" ::: "memory");
        __syncthreads();
        if (it + 1 < nSteps) stage(cur ^ 1, (it + 1) * 32);

        // A fragments: lane<16 -> K {0..7,16..23}; lane>=16 -> {8..15,24..31}
        FragAB a[2];
        #pragma unroll
        for (int mi = 0; mi < 2; ++mi) {
            const unsigned short* pa =
                &ldsX[cur][(wr * 32 + mi * 16 + (lane & 15)) * XS + hi * 8];
            a[mi].q[0] = *(const v4u*)pa;
            a[mi].q[1] = *(const v4u*)(pa + 16);
        }
        #pragma unroll
        for (int j = 0; j < 4; ++j) {
            // B fragment: lane<16 -> K {0..7,16..23}; lane>=16 -> {8..15,24..31}
            FragAB bf;
            const unsigned short* pb =
                &ldsW[cur][(wc * 64 + j * 16 + (lane & 15)) * XS + hi * 8];
            bf.q[0] = *(const v4u*)pb;
            bf.q[1] = *(const v4u*)(pb + 16);
            #pragma unroll
            for (int mi = 0; mi < 2; ++mi)
                acc[mi][j] = __builtin_amdgcn_wmma_f32_16x16x32_bf16(
                    false, a[mi].v, false, bf.v, (short)0, acc[mi][j], false, false);
        }
    }

    // ---- epilogue -----------------------------------------------------------
    #pragma unroll
    for (int mi = 0; mi < 2; ++mi) {
        #pragma unroll
        for (int j = 0; j < 4; ++j) {
            if constexpr (OUT_MODE == OUT_VT) {
                // Pack 8 accumulator rows (8 contiguous tokens) into one b128;
                // store transposed per head: Vt[(b*H+h)*DK+dl][s].
                union { v4u q; unsigned short s[8]; } pk;
                #pragma unroll
                for (int r = 0; r < 8; ++r) pk.s[r] = f2bf(acc[mi][j][r]);
                int dcol = tN + wc * 64 + j * 16 + (lane & 15);
                int hIdx = dcol >> 6, dl = dcol & 63;
                int tok0 = tM + wr * 32 + mi * 16 + hi * 8;
                int bIdx = tok0 / kS, s0 = tok0 % kS;
                unsigned short* Ou = (unsigned short*)Ov;
                *(v4u*)(Ou + ((size_t)(bIdx * kH + hIdx) * kDK + dl) * kS + s0)
                    = pk.q;
            } else {
                #pragma unroll
                for (int r = 0; r < 8; ++r) {
                    int row = tM + wr * 32 + mi * 16 + (hi ? r + 8 : r);
                    int col = tN + wc * 64 + j * 16 + (lane & 15);
                    if constexpr (OUT_MODE == OUT_F32NT)
                        __builtin_nontemporal_store(acc[mi][j][r],
                            (float*)Ov + (size_t)row * N + col);
                    else
                        ((unsigned short*)Ov)[(size_t)row * N + col]
                            = f2bf(acc[mi][j][r]);
                }
            }
        }
    }
}

// ---------------------------------------------------------------------------
// Attention: one workgroup per (b, h, 16-row q tile). 128 threads = 4 waves.
//   phase 1: scores 16x2048 via WMMA; K-fragments load straight from global
//            (K rows contiguous in d == B-fragment layout). Mask via NT loads.
//   phase 2: per-row softmax; normalized attn -> d_out with NT stores (512MB
//            streaming, keep it out of L2) and -> LDS as bf16 for phase 3.
//   phase 3: x = attn @ V with V pre-transposed (Vt); zero barriers, B-frags
//            are contiguous global loads that stay L2-resident.
// ---------------------------------------------------------------------------
__global__ __launch_bounds__(128)
void attention16(const unsigned short* __restrict__ Qbf,
                 const unsigned short* __restrict__ Kbf,
                 const unsigned short* __restrict__ Vt,
                 const int*            __restrict__ mask,
                 float*                __restrict__ attnOut,
                 unsigned short*       __restrict__ Xbf)
{
    constexpr int SSTR = kS + 8;    // padded f32 row stride (scores/exp)
    constexpr int BSTR = kS + 16;   // padded bf16 row stride (attn copy)
    extern __shared__ char smem[];
    float*          sc = (float*)smem;                          // 16*SSTR f32
    unsigned short* sb = (unsigned short*)(sc + 16 * SSTR);     // 16*BSTR bf16

    const int t    = threadIdx.x;
    const int lane = t & 31;
    const int w    = t >> 5;
    const int hi   = (lane >= 16) ? 1 : 0;

    const int qt = blockIdx.x % (kS / 16);
    const int h  = (blockIdx.x / (kS / 16)) % kH;
    const int b  = blockIdx.x / ((kS / 16) * kH);

    // ---- phase 1: scores = Q Kt / sqrt(DK), masked --------------------------
    FragAB aq[2];                               // Q fragments for d-chunks 0,1
    {
        const int m = lane & 15;
        #pragma unroll
        for (int c = 0; c < 2; ++c) {
            const unsigned short* p =
                Qbf + (size_t)(b * kS + qt * 16 + m) * kD + h * kDK + c * 32 + hi * 8;
            aq[c].q[0] = *(const v4u*)p;
            aq[c].q[1] = *(const v4u*)(p + 16);
        }
    }
    for (int kt = w * 32; kt < w * 32 + 32; ++kt) {   // wave w: 512 keys
        v8f acc = {};
        const int key = kt * 16 + (lane & 15);
        #pragma unroll
        for (int c = 0; c < 2; ++c) {
            FragAB bk;          // B[d][n] = K[n][d]: contiguous in memory
            const unsigned short* p =
                Kbf + (size_t)(b * kS + key) * kD + h * kDK + c * 32 + hi * 8;
            bk.q[0] = *(const v4u*)p;
            bk.q[1] = *(const v4u*)(p + 16);
            acc = __builtin_amdgcn_wmma_f32_16x16x32_bf16(
                      false, aq[c].v, false, bk.v, (short)0, acc, false, false);
        }
        #pragma unroll
        for (int r = 0; r < 8; ++r) {
            int row  = hi ? r + 8 : r;
            int qrow = qt * 16 + row;
            float s  = acc[r] * 0.125f;                 // 1/sqrt(64)
            int mv = __builtin_nontemporal_load(
                         &mask[((size_t)b * kS + qrow) * kS + key]);
            if (mv == 0) s = -1.0e9f;
            sc[row * SSTR + key] = s;
        }
    }
    __syncthreads();

    // ---- phase 2: softmax; attn -> global (NT) and -> LDS bf16 --------------
    for (int rr = w * 4; rr < w * 4 + 4; ++rr) {
        float* srow = sc + rr * SSTR;
        float mx = -3.4e38f;
        for (int c = lane; c < kS; c += 32) mx = fmaxf(mx, srow[c]);
        #pragma unroll
        for (int off = 16; off; off >>= 1) mx = fmaxf(mx, __shfl_xor(mx, off, 32));
        float sum = 0.0f;
        for (int c = lane; c < kS; c += 32) {
            float e = __expf(srow[c] - mx);
            srow[c] = e;
            sum += e;
        }
        #pragma unroll
        for (int off = 16; off; off >>= 1) sum += __shfl_xor(sum, off, 32);
        float inv = 1.0f / sum;
        float* gout = attnOut + ((size_t)(b * kH + h) * kS + qt * 16 + rr) * kS;
        for (int c = lane; c < kS; c += 32) {
            float a = srow[c] * inv;
            __builtin_nontemporal_store(a, gout + c);
            sb[rr * BSTR + c] = f2bf(a);
        }
    }
    __syncthreads();

    // ---- phase 3: x = attn @ V (Vt pre-transposed, no barriers) -------------
    v8f xacc = {};
    const unsigned short* vbase =
        Vt + ((size_t)(b * kH + h) * kDK + w * 16 + (lane & 15)) * kS;
    const unsigned short* abase = &sb[(lane & 15) * BSTR];
    #pragma unroll 2
    for (int ch = 0; ch < kS / 32; ++ch) {
        FragAB a;                                   // attn bf16 from LDS
        const unsigned short* pa = abase + ch * 32 + hi * 8;
        a.q[0] = *(const v4u*)pa;
        a.q[1] = *(const v4u*)(pa + 16);
        FragAB bv;                                  // V^T rows, contiguous
        const unsigned short* pv = vbase + ch * 32 + hi * 8;
        bv.q[0] = *(const v4u*)pv;
        bv.q[1] = *(const v4u*)(pv + 16);
        xacc = __builtin_amdgcn_wmma_f32_16x16x32_bf16(
                   false, a.v, false, bv.v, (short)0, xacc, false, false);
    }
    #pragma unroll
    for (int r = 0; r < 8; ++r) {
        int qrow = qt * 16 + (hi ? r + 8 : r);
        int d    = h * kDK + w * 16 + (lane & 15);
        Xbf[(size_t)(b * kS + qrow) * kD + d] = f2bf(xacc[r]);
    }
}

// ---------------------------------------------------------------------------
extern "C" void kernel_launch(void* const* d_in, const int* in_sizes, int n_in,
                              void* d_out, int out_size, void* d_ws, size_t ws_size,
                              hipStream_t stream)
{
    (void)in_sizes; (void)n_in; (void)out_size; (void)ws_size;

    const float* query = (const float*)d_in[0];
    const float* key_  = (const float*)d_in[1];
    const float* value = (const float*)d_in[2];
    const int*   mask  = (const int*)  d_in[3];
    const float* Wq    = (const float*)d_in[4];
    const float* Wk    = (const float*)d_in[5];
    const float* Wv    = (const float*)d_in[6];
    const float* Wo    = (const float*)d_in[7];

    float* out  = (float*)d_out;                       // (B,S,D) f32
    float* attn = out + (size_t)kB * kS * kD;          // (B,H,S,S) f32

    const size_t nTok = (size_t)kB * kS * kD;          // 4M elements
    const size_t nW   = (size_t)kD * kD;               // 1M elements
    unsigned short* p   = (unsigned short*)d_ws;
    unsigned short* qb  = p;  p += nTok;               // bf16 activations
    unsigned short* kb  = p;  p += nTok;
    unsigned short* vb  = p;  p += nTok;
    unsigned short* wqt = p;  p += nW;                 // bf16 transposed weights
    unsigned short* wkt = p;  p += nW;
    unsigned short* wvt = p;  p += nW;
    unsigned short* wot = p;  p += nW;
    unsigned short* Qp  = p;  p += nTok;               // projected Q (B,S,D)
    unsigned short* Kp  = p;  p += nTok;               // projected K (B,S,D)
    unsigned short* Vt  = p;  p += nTok;               // projected V (B,H,DK,S)
    unsigned short* Xb  = p;  p += nTok;               // attn output (B,S,D)

    const int M = kB * kS;                             // 4096

    // prep: activations -> bf16; weights -> bf16 transposed (N x K)
    hipLaunchKernelGGL(cvtBf16, dim3((unsigned)(nTok / (256 * 4)), 1, 3),
                       dim3(256), 0, stream, query, key_, value, qb, kb, vb);
    hipLaunchKernelGGL(transposeW, dim3(kD / 64, kD / 64, 4), dim3(256), 0,
                       stream, Wq, Wk, Wv, Wo, wqt, wkt, wvt, wot);

    dim3 gg(kD / 128, M / 128);                        // (8, 32)
    dim3 bb(256);
    hipLaunchKernelGGL((gemm128<OUT_BF16>),  gg, bb, 0, stream, qb, wqt, (void*)Qp, M, kD, kD);
    hipLaunchKernelGGL((gemm128<OUT_BF16>),  gg, bb, 0, stream, kb, wkt, (void*)Kp, M, kD, kD);
    hipLaunchKernelGGL((gemm128<OUT_VT>),    gg, bb, 0, stream, vb, wvt, (void*)Vt, M, kD, kD);

    const size_t attnLds = (size_t)16 * (kS + 8)  * sizeof(float)
                         + (size_t)16 * (kS + 16) * sizeof(unsigned short);
    (void)hipFuncSetAttribute((const void*)attention16,
                              hipFuncAttributeMaxDynamicSharedMemorySize,
                              (int)attnLds);
    hipLaunchKernelGGL(attention16, dim3(kB * kH * (kS / 16)), dim3(128),
                       attnLds, stream, Qp, Kp, Vt, mask, attn, Xb);

    hipLaunchKernelGGL((gemm128<OUT_F32NT>), gg, bb, 0, stream, Xb, wot, (void*)out, M, kD, kD);
}
